// MoE_5471788335503
// MI455X (gfx1250) — compile-verified, matrix operand order checked by reading, not run on hardware
//
#include <hip/hip_runtime.h>
#include <hip/hip_bf16.h>

// ---------------- problem constants (from reference) ----------------
#define E_NUM 8
#define D_DIM 2048
#define H_DIM 1024
#define TOPK  2
#define T_TOK 2048            // B*L = 2*1024
#define P_MAX (T_TOK * TOPK)  // total token-expert pairs = 4096

#define KC    64              // K-chunk staged in LDS per phase
#define PITCH 72              // KC + 8 bf16 pad -> conflict-free ds_load_b128

#define USE_ASYNC_LDS 1       // gfx1250 GLOBAL_LOAD_ASYNC_TO_LDS path

typedef __attribute__((ext_vector_type(16))) __bf16 v16bf;
typedef __attribute__((ext_vector_type(8)))  float  v8f;

union FragBF {
    uint4  q[2];   // two 16-byte chunks = 16 bf16
    v16bf  v;
};

__device__ __forceinline__ unsigned short f2bf(float f) {
    unsigned int u = __float_as_uint(f);
    unsigned int r = u + 0x7FFFu + ((u >> 16) & 1u);   // round-to-nearest-even
    return (unsigned short)(r >> 16);
}

// ---- async global->LDS copy (16B per lane), tracked by ASYNCcnt ----
__device__ __forceinline__ void cp_g2s_b128(unsigned short* s, const unsigned short* g) {
#if USE_ASYNC_LDS
    unsigned lds_off = (unsigned)(uintptr_t)s;                 // low 32b of generic = LDS offset
    unsigned long long ga = (unsigned long long)(uintptr_t)g;
    asm volatile("global_load_async_to_lds_b128 %0, %1, off"
                 :: "v"(lds_off), "v"(ga) : "memory");
#else
    *(uint4*)s = *(const uint4*)g;
#endif
}
__device__ __forceinline__ void cp_wait() {
#if USE_ASYNC_LDS
    asm volatile("s_wait_asynccnt 0x0" ::: "memory");
#endif
}

// ---------------- fp32 -> bf16 weight conversion ----------------
__global__ __launch_bounds__(256)
void cvt_f32_bf16(const float* __restrict__ src, unsigned short* __restrict__ dst, int n4) {
    int i = blockIdx.x * blockDim.x + threadIdx.x;
    if (i >= n4) return;
    float4 v = ((const float4*)src)[i];
    uint2 packed;
    packed.x = (unsigned int)f2bf(v.x) | ((unsigned int)f2bf(v.y) << 16);
    packed.y = (unsigned int)f2bf(v.z) | ((unsigned int)f2bf(v.w) << 16);
    ((uint2*)dst)[i] = packed;
}

// ---------------- router: logits, top-2, softmax, dispatch lists ----------------
__global__ __launch_bounds__(256)
void moe_router(const float* __restrict__ x,        // [T][D]
                const float* __restrict__ w_gate,   // [E][D]
                unsigned short* __restrict__ x_bf,  // [T][D] bf16 out
                int*   __restrict__ counts,         // [E]
                int*   __restrict__ tok_list,       // [E][T]
                float* __restrict__ gate_list,      // [E][T]
                int*   __restrict__ topk_idx)       // [T][2]
{
    const int wave = threadIdx.x >> 5;
    const int lane = threadIdx.x & 31;
    const int t = blockIdx.x * 8 + wave;

    float acc[E_NUM];
#pragma unroll
    for (int e = 0; e < E_NUM; ++e) acc[e] = 0.0f;

    const float* xrow = x + (size_t)t * D_DIM;
    for (int d = lane; d < D_DIM; d += 32) {
        float xv = xrow[d];
        x_bf[(size_t)t * D_DIM + d] = f2bf(xv);
#pragma unroll
        for (int e = 0; e < E_NUM; ++e)
            acc[e] += xv * w_gate[e * D_DIM + d];
    }
#pragma unroll
    for (int e = 0; e < E_NUM; ++e) {
        float v = acc[e];
        for (int s = 16; s > 0; s >>= 1) v += __shfl_xor(v, s, 32);
        acc[e] = v;
    }
    if (lane == 0) {
        int i0 = 0; float v0 = acc[0];
#pragma unroll
        for (int e = 1; e < E_NUM; ++e) if (acc[e] > v0) { v0 = acc[e]; i0 = e; }
        int i1 = (i0 == 0) ? 1 : 0; float v1 = acc[i1];
#pragma unroll
        for (int e = 0; e < E_NUM; ++e)
            if (e != i0 && acc[e] > v1) { v1 = acc[e]; i1 = e; }
        float e1 = __expf(v1 - v0);
        float inv = 1.0f / (1.0f + e1);
        float g0 = inv, g1 = e1 * inv;

        topk_idx[t * 2 + 0] = i0;
        topk_idx[t * 2 + 1] = i1;

        int s0 = atomicAdd(&counts[i0], 1);
        tok_list[i0 * T_TOK + s0]  = t;
        gate_list[i0 * T_TOK + s0] = g0;
        int s1 = atomicAdd(&counts[i1], 1);
        tok_list[i1 * T_TOK + s1]  = t;
        gate_list[i1 * T_TOK + s1] = g1;
    }
}

// ---------------- compact pairs: offsets + token->pair map ----------------
__global__ __launch_bounds__(256)
void moe_build_pairs(const int*   __restrict__ counts,
                     int*         __restrict__ offsets,     // [E]
                     const int*   __restrict__ tok_list,    // [E][T]
                     const float* __restrict__ gate_list,   // [E][T]
                     const int*   __restrict__ topk_idx,    // [T][2]
                     float*       __restrict__ pair_gate,   // [P]
                     int*         __restrict__ token_pair)  // [T][2]
{
    __shared__ int soff[E_NUM];
    if (threadIdx.x == 0) {
        int run = 0;
        for (int e = 0; e < E_NUM; ++e) { soff[e] = run; offsets[e] = run; run += counts[e]; }
    }
    __syncthreads();
    for (int idx = threadIdx.x; idx < E_NUM * T_TOK; idx += blockDim.x) {
        int e = idx / T_TOK, s = idx % T_TOK;
        if (s < counts[e]) {
            int p = soff[e] + s;
            int t = tok_list[idx];
            pair_gate[p] = gate_list[idx];
            int k = (topk_idx[t * 2 + 0] == e) ? 0 : 1;
            token_pair[t * 2 + k] = p;
        }
    }
}

// ---------------- GEMM1 + GLU: act[p, 0:H] = silu(x@Wi1^T) * (x@Wi2^T) ----------------
// block = 256 thr = 8 waves; tile = 32 rows x 64 h-cols (128 weight rows incl. both halves)
// LDS double-buffered async staging; each wave: one 16x16 C-pair (h1 & h2, A reused)
__global__ __launch_bounds__(256)
void moe_gemm1_glu(const unsigned short* __restrict__ x_bf,     // [T][D]
                   const unsigned short* __restrict__ w_in_bf,  // [E][2H][D]
                   const int*   __restrict__ counts,
                   const int*   __restrict__ offsets,
                   const int*   __restrict__ tok_list,          // [E][T]
                   unsigned short* __restrict__ act_bf)         // [P][H]
{
    const int e   = blockIdx.z;
    const int cnt = counts[e];
    const int m_base = blockIdx.y * 32;
    if (m_base >= cnt) return;
    const int n_base = blockIdx.x * 64;

    const int tid  = threadIdx.x;
    const int wave = tid >> 5;
    const int lane = tid & 31;
    const int mi = wave >> 2;          // 0..1
    const int ni = wave & 3;           // 0..3

    __shared__ unsigned short sA[2][32 * PITCH];    //  9 KB
    __shared__ unsigned short sB[2][128 * PITCH];   // 36 KB

    // ---- per-thread copy descriptors ----
    const int arow = tid >> 3;              // 0..31
    const int achk = (tid & 7) * 8;         // element offset within KC chunk
    int am = m_base + arow; if (am >= cnt) am = cnt - 1;
    const int atok = tok_list[e * T_TOK + am];
    const unsigned short* agp = x_bf + (size_t)atok * D_DIM + achk;
    unsigned short* asp0 = &sA[0][arow * PITCH + achk];
    unsigned short* asp1 = &sA[1][arow * PITCH + achk];

    const unsigned short* bgp[4];
    unsigned short* bsp0[4];
    unsigned short* bsp1[4];
#pragma unroll
    for (int it = 0; it < 4; ++it) {
        int idx = it * 256 + tid;
        int row = idx >> 3;                  // 0..127
        int chk = (idx & 7) * 8;
        int wrow = (row < 64) ? (n_base + row) : (H_DIM + n_base + row - 64);
        bgp[it]  = w_in_bf + ((size_t)e * 2 * H_DIM + wrow) * D_DIM + chk;
        bsp0[it] = &sB[0][row * PITCH + chk];
        bsp1[it] = &sB[1][row * PITCH + chk];
    }

    const int khalf = (lane >> 4) ? 8 : 0;
    const int lrow  = lane & 15;

    v8f c1 = {};
    v8f c2 = {};

    const int NP = D_DIM / KC;   // 32 phases
    // prologue: stage phase 0 into buffer 0
    cp_g2s_b128(asp0, agp);
#pragma unroll
    for (int it = 0; it < 4; ++it) cp_g2s_b128(bsp0[it], bgp[it]);

    for (int p = 0; p < NP; ++p) {
        cp_wait();
        __syncthreads();
        const int buf = p & 1;
        if (p + 1 < NP) {
            const int koff = (p + 1) * KC;
            cp_g2s_b128(buf ? asp0 : asp1, agp + koff);
#pragma unroll
            for (int it = 0; it < 4; ++it)
                cp_g2s_b128(buf ? bsp0[it] : bsp1[it], bgp[it] + koff);
        }
        const unsigned short* sa = sA[buf];
        const unsigned short* sb = sB[buf];
#pragma unroll
        for (int kk = 0; kk < KC; kk += 32) {
            FragBF a, b1, b2;
            const int kl = kk + khalf;
            const unsigned short* ap  = sa + (mi * 16 + lrow) * PITCH + kl;
            const unsigned short* bp1 = sb + (ni * 16 + lrow) * PITCH + kl;
            const unsigned short* bp2 = sb + (64 + ni * 16 + lrow) * PITCH + kl;
            a.q[0]  = *(const uint4*)ap;   a.q[1]  = *(const uint4*)(ap + 16);
            b1.q[0] = *(const uint4*)bp1;  b1.q[1] = *(const uint4*)(bp1 + 16);
            b2.q[0] = *(const uint4*)bp2;  b2.q[1] = *(const uint4*)(bp2 + 16);
            c1 = __builtin_amdgcn_wmma_f32_16x16x32_bf16(false, a.v, false, b1.v, (short)0, c1, false, false);
            c2 = __builtin_amdgcn_wmma_f32_16x16x32_bf16(false, a.v, false, b2.v, (short)0, c2, false, false);
        }
        __syncthreads();
    }

    // ---- fused GLU epilogue ----
    const int off_e  = offsets[e];
    const int hihalf = (lane >> 4) & 1;
    const int m0   = m_base + mi * 16;
    const int ncol = n_base + ni * 16 + lrow;
#pragma unroll
    for (int r = 0; r < 8; ++r) {
        int m = m0 + r + hihalf * 8;
        if (m < cnt) {
            float h1 = c1[r], h2 = c2[r];
            float aval = h1 * (1.0f / (1.0f + __expf(-h1))) * h2;   // silu(h1)*h2
            act_bf[(size_t)(off_e + m) * H_DIM + ncol] = f2bf(aval);
        }
    }
}

// ---------------- GEMM2: contrib[p, 0:D] = gate[p] * (act @ Wout^T) ----------------
// block tile = 32 rows x 128 D-cols; each wave: 2 C tiles sharing the A fragment
__global__ __launch_bounds__(256)
void moe_gemm2(const unsigned short* __restrict__ act_bf,    // [P][H]
               const unsigned short* __restrict__ w_out_bf,  // [E][D][H]
               const int*   __restrict__ counts,
               const int*   __restrict__ offsets,
               const float* __restrict__ pair_gate,          // [P]
               float*       __restrict__ contrib)            // [P][D]
{
    const int e   = blockIdx.z;
    const int cnt = counts[e];
    const int m_base = blockIdx.y * 32;
    if (m_base >= cnt) return;
    const int n_base = blockIdx.x * 128;

    const int tid  = threadIdx.x;
    const int wave = tid >> 5;
    const int lane = tid & 31;
    const int mi = wave >> 2;          // 0..1
    const int ni = wave & 3;           // 0..3
    const int off_e = offsets[e];

    __shared__ unsigned short sA[2][32 * PITCH];
    __shared__ unsigned short sB[2][128 * PITCH];

    const int arow = tid >> 3;
    const int achk = (tid & 7) * 8;
    int am = m_base + arow; if (am >= cnt) am = cnt - 1;
    const unsigned short* agp = act_bf + (size_t)(off_e + am) * H_DIM + achk;
    unsigned short* asp0 = &sA[0][arow * PITCH + achk];
    unsigned short* asp1 = &sA[1][arow * PITCH + achk];

    const unsigned short* bgp[4];
    unsigned short* bsp0[4];
    unsigned short* bsp1[4];
#pragma unroll
    for (int it = 0; it < 4; ++it) {
        int idx = it * 256 + tid;
        int row = idx >> 3;                  // 0..127
        int chk = (idx & 7) * 8;
        bgp[it]  = w_out_bf + ((size_t)e * D_DIM + n_base + row) * H_DIM + chk;
        bsp0[it] = &sB[0][row * PITCH + chk];
        bsp1[it] = &sB[1][row * PITCH + chk];
    }

    const int khalf = (lane >> 4) ? 8 : 0;
    const int lrow  = lane & 15;

    v8f cA = {};
    v8f cB = {};

    const int NP = H_DIM / KC;   // 16 phases
    cp_g2s_b128(asp0, agp);
#pragma unroll
    for (int it = 0; it < 4; ++it) cp_g2s_b128(bsp0[it], bgp[it]);

    for (int p = 0; p < NP; ++p) {
        cp_wait();
        __syncthreads();
        const int buf = p & 1;
        if (p + 1 < NP) {
            const int koff = (p + 1) * KC;
            cp_g2s_b128(buf ? asp0 : asp1, agp + koff);
#pragma unroll
            for (int it = 0; it < 4; ++it)
                cp_g2s_b128(buf ? bsp0[it] : bsp1[it], bgp[it] + koff);
        }
        const unsigned short* sa = sA[buf];
        const unsigned short* sb = sB[buf];
#pragma unroll
        for (int kk = 0; kk < KC; kk += 32) {
            FragBF a, bA, bB;
            const int kl = kk + khalf;
            const unsigned short* ap  = sa + (mi * 16 + lrow) * PITCH + kl;
            const unsigned short* bpA = sb + (ni * 32 + lrow) * PITCH + kl;
            const unsigned short* bpB = sb + (ni * 32 + 16 + lrow) * PITCH + kl;
            a.q[0]  = *(const uint4*)ap;   a.q[1]  = *(const uint4*)(ap + 16);
            bA.q[0] = *(const uint4*)bpA;  bA.q[1] = *(const uint4*)(bpA + 16);
            bB.q[0] = *(const uint4*)bpB;  bB.q[1] = *(const uint4*)(bpB + 16);
            cA = __builtin_amdgcn_wmma_f32_16x16x32_bf16(false, a.v, false, bA.v, (short)0, cA, false, false);
            cB = __builtin_amdgcn_wmma_f32_16x16x32_bf16(false, a.v, false, bB.v, (short)0, cB, false, false);
        }
        __syncthreads();
    }

    const int hihalf = (lane >> 4) & 1;
    const int m0 = m_base + mi * 16;
    const int ncA = n_base + ni * 32 + lrow;
    const int ncB = ncA + 16;
#pragma unroll
    for (int r = 0; r < 8; ++r) {
        int m = m0 + r + hihalf * 8;
        if (m < cnt) {
            int p = off_e + m;
            float g = pair_gate[p];
            contrib[(size_t)p * D_DIM + ncA] = cA[r] * g;
            contrib[(size_t)p * D_DIM + ncB] = cB[r] * g;
        }
    }
}

// ---------------- combine: y[t] = contrib[p0] + contrib[p1] + bias ----------------
__global__ __launch_bounds__(256)
void moe_combine(const float* __restrict__ contrib,     // [P][D]
                 const int*   __restrict__ token_pair,  // [T][2]
                 const float* __restrict__ bias,        // [D]
                 float*       __restrict__ out)         // [T*D] then loss
{
    const int t  = blockIdx.x;
    const int p0 = token_pair[t * 2 + 0];
    const int p1 = token_pair[t * 2 + 1];
    const float4* c0 = (const float4*)(contrib + (size_t)p0 * D_DIM);
    const float4* c1 = (const float4*)(contrib + (size_t)p1 * D_DIM);
    const float4* bz = (const float4*)bias;
    float4* o = (float4*)(out + (size_t)t * D_DIM);
    for (int i = threadIdx.x; i < D_DIM / 4; i += blockDim.x) {
        float4 a = c0[i], b = c1[i], bb = bz[i];
        float4 r;
        r.x = a.x + b.x + bb.x;
        r.y = a.y + b.y + bb.y;
        r.z = a.z + b.z + bb.z;
        r.w = a.w + b.w + bb.w;
        o[i] = r;
    }
    if (t == 0 && threadIdx.x == 0)
        out[(size_t)T_TOK * D_DIM] = 0.0f;   // router aux loss (eval mode) = 0
}

// ---------------- host launch ----------------
extern "C" void kernel_launch(void* const* d_in, const int* in_sizes, int n_in,
                              void* d_out, int out_size, void* d_ws, size_t ws_size,
                              hipStream_t stream) {
    const float* x      = (const float*)d_in[0];  // [B,L,D]
    const float* w_gate = (const float*)d_in[1];  // [E,D]
    const float* w_in   = (const float*)d_in[2];  // [E,2H,D]
    const float* w_out  = (const float*)d_in[3];  // [E,D,H]
    const float* bias   = (const float*)d_in[4];  // [D]
    float* out = (float*)d_out;

    char* ws = (char*)d_ws;
    size_t o = 0;
    auto alloc = [&](size_t bytes) -> void* {
        o = (o + 255) & ~(size_t)255;
        void* p = ws + o;
        o += bytes;
        return p;
    };
    unsigned short* x_bf     = (unsigned short*)alloc((size_t)T_TOK * D_DIM * 2);             // 8 MB
    unsigned short* w_in_bf  = (unsigned short*)alloc((size_t)E_NUM * 2 * H_DIM * D_DIM * 2); // 64 MB
    unsigned short* w_out_bf = (unsigned short*)alloc((size_t)E_NUM * D_DIM * H_DIM * 2);     // 32 MB
    unsigned short* act_bf   = (unsigned short*)alloc((size_t)P_MAX * H_DIM * 2);             // 8 MB
    float*          contrib  = (float*)alloc((size_t)P_MAX * D_DIM * 4);                      // 32 MB
    int*   counts     = (int*)alloc(E_NUM * 4);
    int*   offsets    = (int*)alloc(E_NUM * 4);
    int*   tok_list   = (int*)alloc((size_t)E_NUM * T_TOK * 4);
    float* gate_list  = (float*)alloc((size_t)E_NUM * T_TOK * 4);
    int*   topk_idx   = (int*)alloc((size_t)T_TOK * TOPK * 4);
    float* pair_gate  = (float*)alloc((size_t)P_MAX * 4);
    int*   token_pair = (int*)alloc((size_t)T_TOK * TOPK * 4);

    hipMemsetAsync(counts, 0, E_NUM * sizeof(int), stream);

    // 1) weights fp32 -> bf16
    {
        int n4_in  = E_NUM * 2 * H_DIM * D_DIM / 4;
        int n4_out = E_NUM * D_DIM * H_DIM / 4;
        cvt_f32_bf16<<<(n4_in  + 255) / 256, 256, 0, stream>>>(w_in,  w_in_bf,  n4_in);
        cvt_f32_bf16<<<(n4_out + 255) / 256, 256, 0, stream>>>(w_out, w_out_bf, n4_out);
    }
    // 2) router (also emits x in bf16)
    moe_router<<<T_TOK / 8, 256, 0, stream>>>(x, w_gate, x_bf, counts, tok_list, gate_list, topk_idx);
    // 3) compact pairs
    moe_build_pairs<<<1, 256, 0, stream>>>(counts, offsets, tok_list, gate_list, topk_idx,
                                           pair_gate, token_pair);
    // 4) GEMM1 + GLU  (grid: H cols / 64, worst-case rows per expert, experts)
    moe_gemm1_glu<<<dim3(H_DIM / 64, T_TOK / 32, E_NUM), 256, 0, stream>>>(
        x_bf, w_in_bf, counts, offsets, tok_list, act_bf);
    // 5) GEMM2 (scaled by gate; grid: D cols / 128)
    moe_gemm2<<<dim3(D_DIM / 128, T_TOK / 32, E_NUM), 256, 0, stream>>>(
        act_bf, w_out_bf, counts, offsets, pair_gate, contrib);
    // 6) combine + loss
    moe_combine<<<T_TOK, 256, 0, stream>>>(contrib, token_pair, bias, out);
}
